// QuantLinear_27109833572633
// MI455X (gfx1250) — matrix operand order repaired and checked by reading, not compiled, not run on hardware
//
#include <hip/hip_runtime.h>

typedef _Float16 v16h __attribute__((ext_vector_type(16)));
typedef _Float16 v8h  __attribute__((ext_vector_type(8)));
typedef _Float16 h2   __attribute__((ext_vector_type(2)));
typedef float    v8f  __attribute__((ext_vector_type(8)));
typedef float    f32x4 __attribute__((ext_vector_type(4)));
typedef unsigned int u32x4 __attribute__((ext_vector_type(4)));

#define IN_F   4096
#define OUT_F  11008
#define TOKENS 4096
#define BK     32
#define BM     128
#define BN     256
#define LDA    40   // padded LDS row stride (halves): 80B keeps 16-lane b128 reads spread
#define LDB    40

// ---------------------------------------------------------------------------
// Pre-pass: xsum[m] = sum_k x[m,k]   (factored zero-point term)
// ---------------------------------------------------------------------------
__global__ void rowsum_kernel(const float* __restrict__ x, float* __restrict__ xs) {
    __shared__ float red[256];
    const int m = blockIdx.x;
    const float* row = x + (size_t)m * IN_F;
    float s = 0.f;
    for (int k = threadIdx.x; k < IN_F; k += 256) s += row[k];
    red[threadIdx.x] = s;
    __syncthreads();
    for (int off = 128; off > 0; off >>= 1) {
        if (threadIdx.x < off) red[threadIdx.x] += red[threadIdx.x + off];
        __syncthreads();
    }
    if (threadIdx.x == 0) xs[m] = red[0];
}

// ---------------------------------------------------------------------------
// Main WMMA GEMM: y[m,n] = s[n]*sum_k x_f16[m,k]*q[k,n] - z[n]*xsum[m] + b[n]
// Block: 256 thr (8 wave32), tile 128x256, wave tile 64x64, K-step 32,
// double-buffered LDS, one barrier per K-iteration.
// ---------------------------------------------------------------------------
__global__ __launch_bounds__(256)
void qgemm_wmma_kernel(const float* __restrict__ x,
                       const unsigned int* __restrict__ qw,
                       const float* __restrict__ scales,
                       const float* __restrict__ zeros,
                       const float* __restrict__ bias,
                       const float* __restrict__ xsum,
                       float* __restrict__ out)
{
    __shared__ _Float16 Abuf[2][BM * LDA];   // [m][k] f16, padded
    __shared__ _Float16 Bbuf[2][BN * LDB];   // [n][k] f16 (transposed), padded

    const int tid  = threadIdx.x;
    const int lane = tid & 31;
    const int wave = tid >> 5;
    const int wm   = wave & 1;     // 2 waves along M, 64 rows each
    const int wn   = wave >> 1;    // 4 waves along N, 64 cols each
    const int l16  = lane & 15;
    const int half = lane >> 4;    // 0: lanes 0-15, 1: lanes 16-31

    const int mbase = blockIdx.y * BM;
    const int nbase = blockIdx.x * BN;

    // ---- staging work distribution ----
    // A: thread -> (row 0..127, k-half 0/1): 16 consecutive f32 -> 16 f16
    const int ar  = tid >> 1;
    const int akh = tid & 1;
    // B: thread -> column (0..255): 4 packed int32 words (k=0..31) -> 32 f16
    const int bn  = tid;

    const float*        aG = x  + (size_t)(mbase + ar) * IN_F + akh * 16;
    const unsigned int* bG = qw + (nbase + bn);

    f32x4        areg[4];
    unsigned int breg[4];

    auto glb_load = [&](int kt) {
        const f32x4* ap = (const f32x4*)(aG + kt * BK);
        areg[0] = ap[0]; areg[1] = ap[1]; areg[2] = ap[2]; areg[3] = ap[3];
        const unsigned int* bp = bG + (size_t)(kt * (BK / 8)) * OUT_F;
        #pragma unroll
        for (int j = 0; j < 4; ++j) breg[j] = bp[(size_t)j * OUT_F];
    };

    auto lds_write = [&](int buf) {
        // ---- A: cvt f32 -> f16, pack pairs, one 32B contiguous store ----
        union { unsigned int u; h2 h; } ac[8];
        #pragma unroll
        for (int i = 0; i < 8; ++i) {
            h2 p;
            p[0] = (_Float16)areg[i >> 1][(i & 1) * 2 + 0];
            p[1] = (_Float16)areg[i >> 1][(i & 1) * 2 + 1];
            ac[i].h = p;
        }
        u32x4* adst = (u32x4*)&Abuf[buf][ar * LDA + akh * 16];
        adst[0] = (u32x4){ac[0].u, ac[1].u, ac[2].u, ac[3].u};
        adst[1] = (u32x4){ac[4].u, ac[5].u, ac[6].u, ac[7].u};

        // ---- B: nibble -> f16 via 1024.0 mantissa trick, N-major store ----
        const h2 k1024 = {(_Float16)1024.0f, (_Float16)1024.0f};
        union { unsigned int u; h2 h; } bc[16];
        #pragma unroll
        for (int w = 0; w < 4; ++w) {
            const unsigned int w32 = breg[w];
            #pragma unroll
            for (int p = 0; p < 4; ++p) {
                unsigned int lo = (w32 >> (8 * p)) & 0xFu;
                unsigned int hi = (w32 >> (8 * p + 4)) & 0xFu;
                union { unsigned int u; h2 h; } t;
                t.u = lo | (hi << 16) | 0x64006400u;   // (1024+lo, 1024+hi)
                bc[w * 4 + p].h = t.h - k1024;         // exact int in f16
            }
        }
        u32x4* bdst = (u32x4*)&Bbuf[buf][bn * LDB];
        #pragma unroll
        for (int q = 0; q < 4; ++q)
            bdst[q] = (u32x4){bc[q*4+0].u, bc[q*4+1].u, bc[q*4+2].u, bc[q*4+3].u};
    };

    v8f acc[4][4] = {};

    auto compute = [&](int buf) {
        const _Float16* As = Abuf[buf];
        const _Float16* Bs = Bbuf[buf];
        // A fragments: lanes 0-15 hold K=0..7 & 16..23, lanes 16-31 K=8..15 & 24..31
        v16h afrag[4];
        #pragma unroll
        for (int mt = 0; mt < 4; ++mt) {
            const int row = wm * 64 + mt * 16 + l16;
            v8h lo = *(const v8h*)&As[row * LDA + half * 8];
            v8h hi = *(const v8h*)&As[row * LDA + 16 + half * 8];
            afrag[mt] = __builtin_shufflevector(lo, hi,
                0,1,2,3,4,5,6,7,8,9,10,11,12,13,14,15);
        }
        #pragma unroll
        for (int nt = 0; nt < 4; ++nt) {
            const int nr = wn * 64 + nt * 16 + l16;
            // B fragment: lane holds column nr, K = half*8..half*8+7 & 16+half*8..16+half*8+7
            v8h lo = *(const v8h*)&Bs[nr * LDB + half * 8];
            v8h hi = *(const v8h*)&Bs[nr * LDB + 16 + half * 8];
            v16h bfrag = __builtin_shufflevector(lo, hi,
                0,1,2,3,4,5,6,7,8,9,10,11,12,13,14,15);
            #pragma unroll
            for (int mt = 0; mt < 4; ++mt) {
                acc[mt][nt] = __builtin_amdgcn_wmma_f32_16x16x32_f16(
                    false, afrag[mt], false, bfrag,
                    (short)0, acc[mt][nt], false, false);
            }
        }
    };

    // ---- main loop: single barrier per K-iteration, reg prefetch dist 1,
    //      L2 prefetch dist 2 (global_prefetch_b8) ----
    const int NK = IN_F / BK;   // 128
    glb_load(0);
    for (int kt = 0; kt < NK; ++kt) {
        lds_write(kt & 1);
        __syncthreads();
        if (kt + 1 < NK) glb_load(kt + 1);
        if (kt + 2 < NK) {
            __builtin_prefetch(aG + (kt + 2) * BK, 0, 0);
            __builtin_prefetch(bG + (size_t)((kt + 2) * (BK / 8)) * OUT_F, 0, 0);
        }
        compute(kt & 1);
        // next iteration writes the *other* buffer; the barrier above already
        // fenced every wave's reads of it (last touched two iterations ago)
    }

    // ---- epilogue: y = s*acc - z*xsum + bias ----
    float xs[4][8];
    #pragma unroll
    for (int mt = 0; mt < 4; ++mt)
        #pragma unroll
        for (int r = 0; r < 8; ++r)
            xs[mt][r] = xsum[mbase + wm * 64 + mt * 16 + half * 8 + r];

    #pragma unroll
    for (int nt = 0; nt < 4; ++nt) {
        const int N = nbase + wn * 64 + nt * 16 + l16;
        const float s = scales[N];
        const float z = zeros[N];
        const float b = bias[N];
        #pragma unroll
        for (int mt = 0; mt < 4; ++mt) {
            const int Mb = mbase + wm * 64 + mt * 16 + half * 8;
            #pragma unroll
            for (int r = 0; r < 8; ++r) {
                out[(size_t)(Mb + r) * OUT_F + N] =
                    s * acc[mt][nt][r] - z * xs[mt][r] + b;
            }
        }
    }
}

// ---------------------------------------------------------------------------
extern "C" void kernel_launch(void* const* d_in, const int* in_sizes, int n_in,
                              void* d_out, int out_size, void* d_ws, size_t ws_size,
                              hipStream_t stream) {
    const float*        x       = (const float*)d_in[0];
    const unsigned int* qweight = (const unsigned int*)d_in[1];
    const float*        scales  = (const float*)d_in[2];
    const float*        zeros   = (const float*)d_in[3];
    const float*        bias    = (const float*)d_in[4];
    float*              out     = (float*)d_out;
    float*              xsum    = (float*)d_ws;   // TOKENS floats = 16 KB

    rowsum_kernel<<<TOKENS, 256, 0, stream>>>(x, xsum);

    dim3 grid(OUT_F / BN, TOKENS / BM);   // 43 x 32 = 1376 blocks
    qgemm_wmma_kernel<<<grid, 256, 0, stream>>>(
        x, qweight, scales, zeros, bias, xsum, out);
}
